// GraphAttentionNetwork_30666066493847
// MI455X (gfx1250) — matrix operand order repaired and checked by reading, not compiled, not run on hardware
//
#include <hip/hip_runtime.h>
#include <math.h>

typedef __attribute__((ext_vector_type(16))) _Float16 v16h;
typedef __attribute__((ext_vector_type(8)))  float    v8f;

#define HD    256      // H*D
#define HEADS 4
#define DH    64
#define NOUTC 128
#define SLOPE 0.2f
#define MBLK  5        // M-tiles per wave in GEMM; (30000/16) % 5 == 0
#define CDIV(a,b) (((a)+(b)-1)/(b))

__device__ __forceinline__ float gelu_exact(float x) {
    return 0.5f * x * (1.0f + erff(x * 0.70710678118654752f));
}

__device__ __forceinline__ void atomicMaxF(float* addr, float val) {
    if (val >= 0.0f) atomicMax((int*)addr, __float_as_int(val));
    else             atomicMin((unsigned int*)addr, __float_as_uint(val));
}
__device__ __forceinline__ void atomicMinF(float* addr, float val) {
    if (val >= 0.0f) atomicMin((int*)addr, __float_as_int(val));
    else             atomicMax((unsigned int*)addr, __float_as_uint(val));
}

__device__ __forceinline__ void edge_sd(const int* __restrict__ ei, int E, int e,
                                        int& s, int& d) {
    if (e < E) { s = ei[e]; d = ei[E + e]; }
    else       { s = e - E; d = e - E; }        // self loops appended
}

// ---------------------------------------------------------------------------
// Pack activations [M, Ksrc] (f32, optional GELU) into A-fragment-ordered f16.
// A-matrix 16x32 f16 layout per ISA: lane = (m&15) + 16*((kk&15)>=8),
//                                    i    = (kk>>4)*8 + (kk&7),  kk = k & 31.
// v16h element index = ((m>>4)*nk + kt)*32 + lane. Zero-pads k in [Ksrc,Kpad).
// ---------------------------------------------------------------------------
__global__ void pack_act_kernel(const float* __restrict__ h, _Float16* __restrict__ ap,
                                int M, int Kpad, int Ksrc, int applyGelu) {
    int idx = blockIdx.x * blockDim.x + threadIdx.x;
    int total = M * Kpad;
    if (idx >= total) return;
    int m = idx / Kpad, k = idx - m * Kpad;
    float v = 0.0f;
    if (k < Ksrc) {
        v = h[(size_t)m * Ksrc + k];
        if (applyGelu) v = gelu_exact(v);
    }
    int nk = Kpad >> 5;
    int kt = k >> 5, kk = k & 31;
    int lane = (m & 15) + (((kk & 15) >= 8) ? 16 : 0);
    int i    = ((kk >> 4) << 3) + (kk & 7);
    size_t off = ((size_t)((m >> 4) * nk + kt) * 32 + lane) * 16 + i;
    ap[off] = (_Float16)v;
}

// ---------------------------------------------------------------------------
// Pack weights [Ksrc, Ncols] f32 into B-fragment-ordered f16 (zero pad K).
// B-matrix 32x16 f16 layout: lane = (n&15) + 16*((kk&15)>=8), i = (kk>>4)*8 + (kk&7).
// v16h element index = (kt*(Ncols>>4) + (n>>4))*32 + lane.
// ---------------------------------------------------------------------------
__global__ void pack_w_kernel(const float* __restrict__ W, _Float16* __restrict__ bp,
                              int Kpad, int Ksrc, int Ncols) {
    int idx = blockIdx.x * blockDim.x + threadIdx.x;
    int total = Kpad * Ncols;
    if (idx >= total) return;
    int k = idx / Ncols, n = idx - k * Ncols;
    float v = (k < Ksrc) ? W[(size_t)k * Ncols + n] : 0.0f;
    int kt = k >> 5, kk = k & 31;
    int lane = (n & 15) + (((kk & 15) >= 8) ? 16 : 0);
    int i    = ((kk >> 4) << 3) + (kk & 7);
    size_t off = ((size_t)(kt * (Ncols >> 4) + (n >> 4)) * 32 + lane) * 16 + i;
    bp[off] = (_Float16)v;
}

// ---------------------------------------------------------------------------
// WMMA GEMM: C[M,Ncols] = A(f16 packed) x B(f16 packed), f32 accumulate.
// One wave per (MBLK x 1) group of 16x16 tiles: B fragment loaded once per
// K-step and reused across MBLK WMMAs -> 5x less B traffic, higher
// wmma density per vmem op. Requires (M/16) % MBLK == 0.
// ---------------------------------------------------------------------------
__global__ void gemm_wmma_kernel(const _Float16* __restrict__ ap,
                                 const _Float16* __restrict__ bp,
                                 float* __restrict__ C,
                                 int M, int K, int Ncols) {
    int wave = (blockIdx.x * blockDim.x + threadIdx.x) >> 5;
    int lane = threadIdx.x & 31;
    int ntn = Ncols >> 4;
    int ngrp = (M >> 4) / MBLK;
    if (wave >= ngrp * ntn) return;
    int g  = wave / ntn;
    int tn = wave - g * ntn;
    int tm0 = g * MBLK;
    int nk = K >> 5;

    const v16h* Af = (const v16h*)ap + (size_t)tm0 * nk * 32 + lane;
    const v16h* Bf = (const v16h*)bp + (size_t)tn * 32 + lane;

    v8f acc[MBLK];
    #pragma unroll
    for (int mb = 0; mb < MBLK; ++mb) acc[mb] = (v8f){};

    for (int kt = 0; kt < nk; ++kt) {
        v16h b = Bf[(size_t)kt * ntn * 32];
        #pragma unroll
        for (int mb = 0; mb < MBLK; ++mb) {
            v16h a = Af[((size_t)mb * nk + kt) * 32];
            acc[mb] = __builtin_amdgcn_wmma_f32_16x16x32_f16(
                /*neg_a=*/false, a, /*neg_b=*/false, b,
                /*c_mod=*/(short)0, acc[mb], /*reuse_a=*/false, /*reuse_b=*/false);
        }
    }

    int col = tn * 16 + (lane & 15);
    int rb  = (lane >> 4) << 3;
    #pragma unroll
    for (int mb = 0; mb < MBLK; ++mb) {
        int rbase = (tm0 + mb) * 16 + rb;
        #pragma unroll
        for (int r = 0; r < 8; ++r)
            C[(size_t)(rbase + r) * Ncols + col] = acc[mb][r];
    }
}

// alpha_src/alpha_dst: per (node, head) dot of h[n,h,:] with attention vectors
__global__ void alpha_kernel(const float* __restrict__ h,
                             const float* __restrict__ a_src,
                             const float* __restrict__ a_dst,
                             float* __restrict__ as, float* __restrict__ ad, int N) {
    int idx = blockIdx.x * blockDim.x + threadIdx.x;
    if (idx >= N * HEADS) return;
    int n = idx >> 2, hh = idx & 3;
    const float* hp = h + (size_t)n * HD + hh * DH;
    const float* sv = a_src + hh * DH;
    const float* dv = a_dst + hh * DH;
    float accs = 0.0f, accd = 0.0f;
    for (int d = 0; d < DH; ++d) { float v = hp[d]; accs += v * sv[d]; accd += v * dv[d]; }
    as[idx] = accs; ad[idx] = accd;
}

__global__ void init_softmax_kernel(float* __restrict__ nmax, float* __restrict__ nsum, int N) {
    int idx = blockIdx.x * blockDim.x + threadIdx.x;
    if (idx >= N * HEADS) return;
    nmax[idx] = -INFINITY;
    nsum[idx] = 0.0f;
}

__global__ void init_out_kernel(float* __restrict__ out, const float* __restrict__ bias, int N) {
    int idx = blockIdx.x * blockDim.x + threadIdx.x;
    if (idx >= N * HD) return;
    out[idx] = bias[idx & (HD - 1)];
}

__global__ void edge_max_kernel(const int* __restrict__ ei, int E, int Etot,
                                const float* __restrict__ as, const float* __restrict__ ad,
                                float* __restrict__ nmax) {
    int idx = blockIdx.x * blockDim.x + threadIdx.x;
    if (idx >= Etot * HEADS) return;
    int e = idx >> 2, hh = idx & 3;
    int s, d; edge_sd(ei, E, e, s, d);
    float v = as[s * HEADS + hh] + ad[d * HEADS + hh];
    v = (v > 0.0f) ? v : SLOPE * v;
    atomicMaxF(&nmax[d * HEADS + hh], v);
}

__global__ void edge_expsum_kernel(const int* __restrict__ ei, int E, int Etot,
                                   const float* __restrict__ as, const float* __restrict__ ad,
                                   const float* __restrict__ nmax,
                                   float* __restrict__ ew, float* __restrict__ nsum) {
    int idx = blockIdx.x * blockDim.x + threadIdx.x;
    if (idx >= Etot * HEADS) return;
    int e = idx >> 2, hh = idx & 3;
    int s, d; edge_sd(ei, E, e, s, d);
    float v = as[s * HEADS + hh] + ad[d * HEADS + hh];
    v = (v > 0.0f) ? v : SLOPE * v;
    float w = __expf(v - nmax[d * HEADS + hh]);
    ew[idx] = w;
    atomicAdd(&nsum[d * HEADS + hh], w);
}

__global__ void aggregate_kernel(const int* __restrict__ ei, int E, int Etot,
                                 const float* __restrict__ h, const float* __restrict__ ew,
                                 const float* __restrict__ nsum, float* __restrict__ out) {
    long total = (long)Etot * HD;
    long stride = (long)gridDim.x * blockDim.x;
    for (long idx = (long)blockIdx.x * blockDim.x + threadIdx.x; idx < total; idx += stride) {
        int e = (int)(idx >> 8);
        int c = (int)(idx & 255);
        int s, d; edge_sd(ei, E, e, s, d);
        int hh = c >> 6;
        float w = ew[e * HEADS + hh] / (nsum[d * HEADS + hh] + 1e-16f);
        atomicAdd(&out[(size_t)d * HD + c], h[(size_t)s * HD + c] * w);
    }
}

// ---------------- readout -----------------
__global__ void init_g_kernel(float* gmin, float* gmax, float* gsum, float* cnt, int B) {
    int idx = blockIdx.x * blockDim.x + threadIdx.x;
    if (idx >= B * HD) return;
    gmin[idx] = INFINITY;
    gmax[idx] = -INFINITY;
    gsum[idx] = 0.0f;
    if (idx < B) cnt[idx] = 0.0f;
}

__global__ void scatter_g_kernel(const float* __restrict__ h, const int* __restrict__ batch,
                                 float* gmin, float* gmax, float* gsum, float* cnt, int N) {
    int idx = blockIdx.x * blockDim.x + threadIdx.x;
    if (idx >= N * HD) return;
    int n = idx >> 8, c = idx & 255;
    int b = batch[n];
    float v = h[idx];
    atomicMinF(&gmin[b * HD + c], v);
    atomicMaxF(&gmax[b * HD + c], v);
    atomicAdd(&gsum[b * HD + c], v);
    if (c == 0) atomicAdd(&cnt[b], 1.0f);
}

__global__ void head_kernel(const float* __restrict__ gmin, const float* __restrict__ gmax,
                            const float* __restrict__ gsum, const float* __restrict__ cnt,
                            const float* __restrict__ head_W, const float* __restrict__ head_b,
                            float* __restrict__ out) {
    int b = blockIdx.x;
    int o = threadIdx.x;
    float c = fmaxf(cnt[b], 1.0f);
    float acc = head_b[o];
    for (int j = 0; j < 4 * HD; ++j) {
        int q = j >> 8, cc = j & 255;
        float gv;
        if      (q == 0) gv = gmin[b * HD + cc];
        else if (q == 1) gv = gmax[b * HD + cc];
        else if (q == 2) gv = gsum[b * HD + cc] / c;
        else             gv = gsum[b * HD + cc];
        acc += gelu_exact(gv) * head_W[(size_t)j * NOUTC + o];
    }
    out[b * NOUTC + o] = acc;
}

// ---------------------------------------------------------------------------
static inline size_t alignup(size_t x) { return (x + 255) & ~(size_t)255; }

extern "C" void kernel_launch(void* const* d_in, const int* in_sizes, int n_in,
                              void* d_out, int out_size, void* d_ws, size_t ws_size,
                              hipStream_t stream) {
    const float* x        = (const float*)d_in[0];
    const int*   ei       = (const int*)  d_in[1];
    const int*   batch    = (const int*)  d_in[2];
    const float* W0       = (const float*)d_in[3];
    const float* a_src0   = (const float*)d_in[4];
    const float* a_dst0   = (const float*)d_in[5];
    const float* b0       = (const float*)d_in[6];
    const float* Ws       = (const float*)d_in[7];
    const float* a_srcs   = (const float*)d_in[8];
    const float* a_dsts   = (const float*)d_in[9];
    const float* bs       = (const float*)d_in[10];
    const float* head_W   = (const float*)d_in[11];
    const float* head_b   = (const float*)d_in[12];
    float* out = (float*)d_out;

    const int N    = in_sizes[2];
    const int E    = in_sizes[1] / 2;
    const int Etot = E + N;
    const int B    = 48;
    const int L    = 10;

    // --- carve workspace ---
    char* p = (char*)d_ws;
    float*     hA    = (float*)p;     p += alignup((size_t)N * HD * 4);
    float*     hB    = (float*)p;     p += alignup((size_t)N * HD * 4);
    _Float16*  apack = (_Float16*)p;  p += alignup((size_t)N * HD * 2);
    _Float16*  bpack = (_Float16*)p;  p += alignup((size_t)HD * HD * 2);
    float*     as    = (float*)p;     p += alignup((size_t)N * HEADS * 4);
    float*     ad    = (float*)p;     p += alignup((size_t)N * HEADS * 4);
    float*     nmax  = (float*)p;     p += alignup((size_t)N * HEADS * 4);
    float*     nsum  = (float*)p;     p += alignup((size_t)N * HEADS * 4);
    float*     ew    = (float*)p;     p += alignup((size_t)Etot * HEADS * 4);
    float*     gmin  = (float*)p;     p += alignup((size_t)B * HD * 4);
    float*     gmax  = (float*)p;     p += alignup((size_t)B * HD * 4);
    float*     gsum  = (float*)p;     p += alignup((size_t)B * HD * 4);
    float*     cnt   = (float*)p;     p += alignup((size_t)B * 4);
    (void)ws_size; (void)n_in; (void)out_size;

    const int TB = 256;

    for (int layer = 0; layer < L; ++layer) {
        int Kpad, Ksrc, gelu_in;
        const float* Wl; const float* asv; const float* adv; const float* bv;
        const float* inAct;
        if (layer == 0) {
            Kpad = 32; Ksrc = 8; gelu_in = 0;
            Wl = W0; asv = a_src0; adv = a_dst0; bv = b0; inAct = x;
        } else {
            Kpad = HD; Ksrc = HD; gelu_in = 1;
            Wl  = Ws     + (size_t)(layer - 1) * HD * HD;
            asv = a_srcs + (size_t)(layer - 1) * HD;
            adv = a_dsts + (size_t)(layer - 1) * HD;
            bv  = bs     + (size_t)(layer - 1) * HD;
            inAct = hB;
        }

        // pack activations (+GELU for layers >= 1) and weights into WMMA order
        pack_act_kernel<<<CDIV(N * Kpad, TB), TB, 0, stream>>>(inAct, apack, N, Kpad, Ksrc, gelu_in);
        pack_w_kernel  <<<CDIV(Kpad * HD, TB), TB, 0, stream>>>(Wl, bpack, Kpad, Ksrc, HD);

        // h = act @ W  (WMMA f16 -> f32), MBLK M-tiles per wave share B frag
        int waves = ((N / 16) / MBLK) * (HD / 16);
        gemm_wmma_kernel<<<CDIV(waves * 32, TB), TB, 0, stream>>>(apack, bpack, hA, N, Kpad, HD);

        // attention coefficients + segment softmax + aggregation into hB
        alpha_kernel<<<CDIV(N * HEADS, TB), TB, 0, stream>>>(hA, asv, adv, as, ad, N);
        init_softmax_kernel<<<CDIV(N * HEADS, TB), TB, 0, stream>>>(nmax, nsum, N);
        edge_max_kernel<<<CDIV(Etot * HEADS, TB), TB, 0, stream>>>(ei, E, Etot, as, ad, nmax);
        edge_expsum_kernel<<<CDIV(Etot * HEADS, TB), TB, 0, stream>>>(ei, E, Etot, as, ad, nmax, ew, nsum);
        init_out_kernel<<<CDIV(N * HD, TB), TB, 0, stream>>>(hB, bv, N);
        aggregate_kernel<<<8192, TB, 0, stream>>>(ei, E, Etot, hA, ew, nsum, hB);
    }

    // readout: segment min/max/mean/sum over graphs, GELU, linear
    init_g_kernel<<<CDIV(B * HD, TB), TB, 0, stream>>>(gmin, gmax, gsum, cnt, B);
    scatter_g_kernel<<<CDIV(N * HD, TB), TB, 0, stream>>>(hB, batch, gmin, gmax, gsum, cnt, N);
    head_kernel<<<B, NOUTC, 0, stream>>>(gmin, gmax, gsum, cnt, head_W, head_b, out);
}
